// HierarchicalDenseLayer_20160576487702
// MI455X (gfx1250) — matrix operand order repaired and checked by reading, not compiled, not run on hardware
//
#include <hip/hip_runtime.h>

#define PI_F 3.1415926f
#define CTXSZ 512
#define NJ 22
#define NCOL 66
#define NPAD 80
#define ROWS_PER_WG 128
#define KSTEP 32
#define NKSTEP (CTXSZ / KSTEP) /* 16 */

typedef __bf16 bf16_t;
typedef __attribute__((ext_vector_type(16))) __bf16 v16bf;
typedef __attribute__((ext_vector_type(8)))  float  v8f;
typedef __attribute__((ext_vector_type(4)))  int    v4i;

// ---------------- workspace layout (bytes) ----------------
#define WS_OFF_BIAS (NPAD * CTXSZ * 2)
#define WS_OFF_WP   (WS_OFF_BIAS + NPAD * 4)

// ---------------- LDS layout (bytes, dynamic) ----------------
#define LDS_OFF_WP  (NPAD * CTXSZ * 2)
#define LDS_OFF_UNI (LDS_OFF_WP + 1024)
#define LDS_TOTAL   (LDS_OFF_UNI + ROWS_PER_WG * NPAD * 4) /* 123904 */

// ---------------- CDNA5 async-to-LDS path (guarded) ----------------
#if defined(__gfx1250__) && __has_builtin(__builtin_amdgcn_global_load_async_to_lds_b128)
#define HAVE_ASYNC_LDS 1
#else
#define HAVE_ASYNC_LDS 0
#endif

#if HAVE_ASYNC_LDS
typedef __attribute__((address_space(1))) v4i* g128_t;  // global 128-bit pointer
typedef __attribute__((address_space(3))) v4i* l128_t;  // LDS 128-bit pointer
static __device__ __forceinline__ g128_t to_gas(const void* p) {
  return (g128_t)(unsigned long long)p;
}
static __device__ __forceinline__ l128_t to_las(void* p) {
  // generic pointers to LDS carry the LDS offset in the low 32 bits
  return (l128_t)(unsigned int)(unsigned long long)p;
}
#endif

static __device__ __forceinline__ void wait_async_zero() {
#if defined(__gfx1250__)
#if __has_builtin(__builtin_amdgcn_s_wait_asynccnt)
  __builtin_amdgcn_s_wait_asynccnt(0);
#elif HAVE_ASYNC_LDS
  asm volatile("s_wait_asynccnt 0x0" ::: "memory");
#endif
#endif
}

// ============================================================
// Pack kernel: build Wcat(bf16), biascat, Wp in workspace.
// ============================================================
__global__ void pack_kernel(const float* __restrict__ W0, const float* __restrict__ b0,
                            const float* __restrict__ W,  const float* __restrict__ b,
                            bf16_t* __restrict__ wcat, float* __restrict__ biascat,
                            float* __restrict__ wp) {
  int idx = blockIdx.x * blockDim.x + threadIdx.x;
  const int total = NPAD * CTXSZ;
  for (int i = idx; i < total; i += gridDim.x * blockDim.x) {
    int n = i / CTXSZ, k = i - n * CTXSZ;
    float v = 0.f;
    if (n < NCOL) {
      int jid = n / 3, c = n - jid * 3;
      v = (jid == 0) ? W0[k * 3 + c] : W[(jid * 515 + k) * 3 + c];
    }
    wcat[n * CTXSZ + k] = (bf16_t)v;
  }
  if (blockIdx.x == 0) {
    int t = threadIdx.x;
    if (t < NPAD) {
      float v = 0.f;
      if (t < NCOL) { int jid = t / 3, c = t - jid * 3; v = (jid == 0) ? b0[c] : b[jid * 3 + c]; }
      biascat[t] = v;
    }
    if (t < NJ * 9) {
      int jid = t / 9, r = (t % 9) / 3, c = t % 3;
      wp[t] = (jid == 0) ? 0.f : W[(jid * 515 + 512 + r) * 3 + c];
    }
  }
}

// ============================================================
// Context chunk staging helpers
// ============================================================
#if HAVE_ASYNC_LDS
__device__ __forceinline__ void issue_chunk_async(const float* __restrict__ ctx, int rowbase,
                                                  int ks, int tid, float* buf) {
  const float* g = ctx + (size_t)(rowbase + (tid >> 1)) * CTXSZ + ks * KSTEP + (tid & 1) * 16;
  float* l = buf + (tid >> 1) * KSTEP + (tid & 1) * 16;
  g128_t gp = to_gas(g);
  l128_t lp = to_las(l);
  __builtin_amdgcn_global_load_async_to_lds_b128(gp, lp, 0, 0);
  __builtin_amdgcn_global_load_async_to_lds_b128(gp, lp, 16, 0);
  __builtin_amdgcn_global_load_async_to_lds_b128(gp, lp, 32, 0);
  __builtin_amdgcn_global_load_async_to_lds_b128(gp, lp, 48, 0);
}
#else
__device__ __forceinline__ void load_chunk(const float* __restrict__ ctx, int rowbase,
                                           int ks, int tid, float4 r[4]) {
  const float* g = ctx + (size_t)(rowbase + (tid >> 1)) * CTXSZ + ks * KSTEP + (tid & 1) * 16;
  r[0] = ((const float4*)g)[0];
  r[1] = ((const float4*)g)[1];
  r[2] = ((const float4*)g)[2];
  r[3] = ((const float4*)g)[3];
}
__device__ __forceinline__ void store_chunk(float* buf, int tid, const float4 r[4]) {
  float* d = buf + (tid >> 1) * KSTEP + (tid & 1) * 16;
  ((float4*)d)[0] = r[0];
  ((float4*)d)[1] = r[1];
  ((float4*)d)[2] = r[2];
  ((float4*)d)[3] = r[3];
}
#endif

// ============================================================
// Main kernel: WMMA GEMM (context @ Wcat + bias) then chain sweep.
// ============================================================
__global__ __launch_bounds__(256)
void hier_kernel(const float* __restrict__ ctx,
                 const bf16_t* __restrict__ wcat_g,
                 const float* __restrict__ bias_g,
                 const float* __restrict__ wp_g,
                 float* __restrict__ out) {
  extern __shared__ char smem[];
  bf16_t* lw   = (bf16_t*)smem;                          // [80][512] bf16
  float*  lwp  = (float*)(smem + LDS_OFF_WP);            // [22*9]
  float*  lctx = (float*)(smem + LDS_OFF_UNI);           // 2 x [128][32]
  float*  ldst = (float*)(smem + LDS_OFF_UNI);           // [128][80] (reuses ctx bufs)

  const int tid  = threadIdx.x;
  const int lane = tid & 31;
  const int wave = tid >> 5;
  const int l15  = lane & 15;
  const int hi   = (lane >> 4) & 1;
  const int rowbase = blockIdx.x * ROWS_PER_WG;
  const int mbase = wave * 16;

  // ---- preload packed weights + parent blocks into LDS ----
  {
    uint4* s = (uint4*)lw;
    const uint4* g = (const uint4*)wcat_g;
    const int n16 = NPAD * CTXSZ * 2 / 16; // 5120, exact multiple of 256
#pragma unroll 4
    for (int i = tid; i < n16; i += 256) s[i] = g[i];
  }
  if (tid < NJ * 9) lwp[tid] = wp_g[tid];

  // ---- accumulators, bias folded in as C init ----
  v8f acc[5];
#pragma unroll
  for (int t = 0; t < 5; ++t) {
    float bv = bias_g[16 * t + l15];
#pragma unroll
    for (int rr = 0; rr < 8; ++rr) acc[t][rr] = bv;
  }

  const int m = mbase + l15; // A-matrix row for this lane

  // ---- GEMM main loop: K = 512 in 16 steps of 32, double-buffered ----
#if HAVE_ASYNC_LDS
  issue_chunk_async(ctx, rowbase, 0, tid, lctx);
  wait_async_zero();
  __syncthreads();

  for (int ks = 0; ks < NKSTEP; ++ks) {
    // prefetch next chunk into the other buffer (async DMA to LDS, overlaps compute)
    if (ks + 1 < NKSTEP)
      issue_chunk_async(ctx, rowbase, ks + 1, tid,
                        lctx + ((ks + 1) & 1) * (ROWS_PER_WG * KSTEP));

    float* cur = lctx + (ks & 1) * (ROWS_PER_WG * KSTEP);
#else
  {
    float4 r[4];
    load_chunk(ctx, rowbase, 0, tid, r);
    store_chunk(lctx, tid, r);
  }
  __syncthreads();

  float4 r[4];
  for (int ks = 0; ks < NKSTEP; ++ks) {
    float* cur = lctx + (ks & 1) * (ROWS_PER_WG * KSTEP);
    if (ks + 1 < NKSTEP) load_chunk(ctx, rowbase, ks + 1, tid, r);
#endif

    // A fragment: 16x32 bf16; lo lanes K {0..15}, hi lanes K {16..31}
    const float* ap = cur + m * KSTEP;
    float4 a0 = *(const float4*)(ap + hi * 16);
    float4 a1 = *(const float4*)(ap + hi * 16 + 4);
    float4 a2 = *(const float4*)(ap + hi * 16 + 8);
    float4 a3 = *(const float4*)(ap + hi * 16 + 12);
    v16bf A;
    A[0]  = (bf16_t)a0.x; A[1]  = (bf16_t)a0.y; A[2]  = (bf16_t)a0.z; A[3]  = (bf16_t)a0.w;
    A[4]  = (bf16_t)a1.x; A[5]  = (bf16_t)a1.y; A[6]  = (bf16_t)a1.z; A[7]  = (bf16_t)a1.w;
    A[8]  = (bf16_t)a2.x; A[9]  = (bf16_t)a2.y; A[10] = (bf16_t)a2.z; A[11] = (bf16_t)a2.w;
    A[12] = (bf16_t)a3.x; A[13] = (bf16_t)a3.y; A[14] = (bf16_t)a3.z; A[15] = (bf16_t)a3.w;

    const int kb = ks * KSTEP;
#pragma unroll
    for (int t = 0; t < 5; ++t) {
      // B fragment: 32x16 bf16; lane n = output col, lo lanes K kb..kb+15, hi lanes kb+16..kb+31
      const bf16_t* wrow = lw + (16 * t + l15) * CTXSZ + kb + hi * 16;
      v16bf Bf = *(const v16bf*)wrow;
      acc[t] = __builtin_amdgcn_wmma_f32_16x16x32_bf16(
          false, A, false, Bf, (short)0, acc[t], false, false);
    }

#if HAVE_ASYNC_LDS
    if (ks + 1 < NKSTEP) wait_async_zero(); // own next-chunk DMAs complete
    __syncthreads();                        // everyone done reading cur / writing next
#else
    if (ks + 1 < NKSTEP) store_chunk(lctx + ((ks + 1) & 1) * (ROWS_PER_WG * KSTEP), tid, r);
    __syncthreads();
#endif
  }

  // ---- drain accumulators to LDS D tile [128][80] ----
#pragma unroll
  for (int t = 0; t < 5; ++t) {
    const int col = 16 * t + l15;
#pragma unroll
    for (int rr = 0; rr < 8; ++rr) {
      ldst[(mbase + hi * 8 + rr) * NPAD + col] = acc[t][rr];
    }
  }
  __syncthreads();

  // ---- sequential kinematic chains: one thread per row ----
  if (tid < ROWS_PER_WG) {
    float* drow = ldst + tid * NPAD;
    float J[NJ][3];
#pragma unroll
    for (int c = 0; c < 3; ++c) J[0][c] = PI_F * tanhf(drow[c]);

    const int par[21]   = {0, 2, 5, 8,  0, 1, 4, 7,  0, 3, 6, 9, 12,  9, 14, 17, 19,  9, 13, 16, 18};
    const int child[21] = {2, 5, 8, 11, 1, 4, 7, 10, 3, 6, 9, 12, 15, 14, 17, 19, 21, 13, 16, 18, 20};
#pragma unroll
    for (int e = 0; e < 21; ++e) {
      const int p = par[e], j = child[e];
#pragma unroll
      for (int c = 0; c < 3; ++c) {
        float v = drow[3 * j + c]
                + J[p][0] * lwp[j * 9 + 0 + c]
                + J[p][1] * lwp[j * 9 + 3 + c]
                + J[p][2] * lwp[j * 9 + 6 + c];
        J[j][c] = PI_F * tanhf(v);
      }
      drow[3 * j + 0] = J[j][0];
      drow[3 * j + 1] = J[j][1];
      drow[3 * j + 2] = J[j][2];
    }
    drow[0] = J[0][0]; drow[1] = J[0][1]; drow[2] = J[0][2];
  }
  __syncthreads();

  // ---- coalesced writeout [128][66] ----
  for (int i = tid; i < ROWS_PER_WG * NCOL; i += 256) {
    int row = i / NCOL, c = i - row * NCOL;
    out[(size_t)(rowbase + row) * NCOL + c] = ldst[row * NPAD + c];
  }
}

// ============================================================
extern "C" void kernel_launch(void* const* d_in, const int* in_sizes, int n_in,
                              void* d_out, int out_size, void* d_ws, size_t ws_size,
                              hipStream_t stream) {
  const float* ctx = (const float*)d_in[0];  // [B, 512]
  const float* W0  = (const float*)d_in[1];  // [512, 3]
  const float* b0  = (const float*)d_in[2];  // [3]
  const float* W   = (const float*)d_in[3];  // [22, 515, 3]
  const float* b   = (const float*)d_in[4];  // [22, 3]
  float* out = (float*)d_out;

  char* ws = (char*)d_ws;
  bf16_t* wcat    = (bf16_t*)ws;
  float*  biascat = (float*)(ws + WS_OFF_BIAS);
  float*  wp      = (float*)(ws + WS_OFF_WP);

  pack_kernel<<<160, 256, 0, stream>>>(W0, b0, W, b, wcat, biascat, wp);

  const int Brows = in_sizes[0] / CTXSZ;       // 131072
  const int nwg = Brows / ROWS_PER_WG;         // 1024
  hier_kernel<<<nwg, 256, LDS_TOTAL, stream>>>(ctx, wcat, biascat, wp, out);
}
